// GT_61400852463745
// MI455X (gfx1250) — matrix operand order repaired and checked by reading, not compiled, not run on hardware
//
#include <hip/hip_runtime.h>
#include <math.h>

// ---------------------------------------------------------------------------
// CDNA5 (gfx1250, wave32) WMMA types
// ---------------------------------------------------------------------------
typedef __attribute__((ext_vector_type(16))) __bf16 v16bf;
typedef __attribute__((ext_vector_type(8)))  float  v8f;
typedef __attribute__((ext_vector_type(4)))  int    v4i;

typedef __attribute__((address_space(1))) v4i* gv4i_p;  // global int4*
typedef __attribute__((address_space(3))) v4i* lv4i_p;  // LDS int4*

union Frag { v16bf v; unsigned int u[8]; };

__device__ __forceinline__ unsigned short f2bf(float f) {
  unsigned int u = __float_as_uint(f);
  return (unsigned short)((u + 0x7FFFu + ((u >> 16) & 1u)) >> 16);
}
__device__ __forceinline__ unsigned int pack2bf(float lo, float hi) {
  return (unsigned int)f2bf(lo) | ((unsigned int)f2bf(hi) << 16);
}

#if __has_builtin(__builtin_amdgcn_global_load_async_to_lds_b128)
#define HAVE_ASYNC_LDS 1
#else
#define HAVE_ASYNC_LDS 0
#endif

// ---------------------------------------------------------------------------
// Generic tiled GEMM:  C[M,N] = act(A_f32[M,K] @ (Bt_bf16[N,K])^T + bias[N])
// Optional second output C2 = C + R (fused residual).
// Block tile: 128(M) x 64(N), K-step 64, 128 threads (4 waves).
// Each wave owns a 32x64 output: 16 WMMAs per K-step from 4 A-frag + 8 B-frag
// ds_load_b128s. LDS rows padded (72/BK=64) for bank-conflict-free b128 reads.
// Bt tiles (pure bf16 copy) staged via GLOBAL_LOAD_ASYNC_TO_LDS_B128 when the
// builtin is available (ASYNCcnt path), else plain LDS stores.
// ---------------------------------------------------------------------------
#define BK   64
#define LDP  72   // padded row stride (ushorts): 144B, multiple of 16B

__global__ __launch_bounds__(128) void gemm_bf16_kernel(
    const float* __restrict__ A, long lda,
    const unsigned short* __restrict__ Bt,
    const float* __restrict__ bias,
    float* __restrict__ C, long ldc,
    float* __restrict__ C2, long ldc2,
    const float* __restrict__ R, long ldr,
    int M, int N, int K, int act)
{
  __shared__ unsigned short As[128 * LDP];  // 18 KB
  __shared__ unsigned short Bs[64 * LDP];   //  9 KB

  const int t    = threadIdx.x;
  const int wave = t >> 5;
  const int lane = t & 31;
  const int half = lane >> 4;
  const int l16  = lane & 15;
  const int m0 = blockIdx.y * 128;
  const int n0 = blockIdx.x * 64;

  v8f acc[2][4] = {};

  for (int k0 = 0; k0 < K; k0 += BK) {
    // ---- stage A tile (f32 -> bf16): row per thread, 64 cols ----
    {
      int m = m0 + t;
      unsigned int* d = (unsigned int*)(As + t * LDP);
      if (m < M) {
        const float* src = A + (long)m * lda + k0;
        if (k0 + 2 * BK <= K) __builtin_prefetch(src + BK, 0, 1);
#pragma unroll
        for (int i = 0; i < 32; ++i) d[i] = pack2bf(src[2 * i], src[2 * i + 1]);
      } else {
#pragma unroll
        for (int i = 0; i < 32; ++i) d[i] = 0u;
      }
    }
    // ---- stage Bt tile (bf16 copy): 64 rows x 64 cols ----
    {
      int n = t >> 1, seg = (t & 1) * 32;  // 32 ushorts = 64 B per thread
      bool valid = (n0 + n) < N;
      const unsigned short* gsrc = Bt + (long)(n0 + n) * K + k0 + seg;
      unsigned short* ldst = Bs + n * LDP + seg;
#if HAVE_ASYNC_LDS
      if (valid) {
#pragma unroll
        for (int i = 0; i < 4; ++i)
          __builtin_amdgcn_global_load_async_to_lds_b128(
              (gv4i_p)(gsrc + i * 8), (lv4i_p)(ldst + i * 8), 0, 0);
      } else {
        unsigned int* d = (unsigned int*)ldst;
#pragma unroll
        for (int i = 0; i < 16; ++i) d[i] = 0u;
      }
      asm volatile("s_wait_asynccnt 0" ::: "memory");
#else
      unsigned int* d = (unsigned int*)ldst;
      if (valid) {
        const unsigned int* s = (const unsigned int*)gsrc;
#pragma unroll
        for (int i = 0; i < 16; ++i) d[i] = s[i];
      } else {
#pragma unroll
        for (int i = 0; i < 16; ++i) d[i] = 0u;
      }
#endif
    }
    __syncthreads();

#pragma unroll
    for (int kk = 0; kk < 2; ++kk) {
      // A fragments: rows wave*32 + mt*16 .. +16
      Frag a[2];
#pragma unroll
      for (int mt = 0; mt < 2; ++mt) {
        const unsigned int* ap =
            (const unsigned int*)(As + (wave * 32 + mt * 16 + l16) * LDP + kk * 32);
#pragma unroll
        for (int j = 0; j < 4; ++j) {
          a[mt].u[j]     = ap[half * 4 + j];      // K = half*8 + 2j
          a[mt].u[4 + j] = ap[8 + half * 4 + j];  // K = 16 + half*8 + 2j
        }
      }
#pragma unroll
      for (int nt = 0; nt < 4; ++nt) {
        Frag b;
        const unsigned int* bp =
            (const unsigned int*)(Bs + (nt * 16 + l16) * LDP + kk * 32);
#pragma unroll
        for (int j = 0; j < 8; ++j) b.u[j] = bp[(j & 4) * 2 + half * 4 + (j & 3)];  // K = (j<4 ? half*8 + 2j : 16 + half*8 + 2(j-4))
        acc[0][nt] = __builtin_amdgcn_wmma_f32_16x16x32_bf16(
            false, a[0].v, false, b.v, (short)0, acc[0][nt], false, false);
        acc[1][nt] = __builtin_amdgcn_wmma_f32_16x16x32_bf16(
            false, a[1].v, false, b.v, (short)0, acc[1][nt], false, false);
      }
    }
    __syncthreads();
  }

  // ---- epilogue: C layout row = rr + 8*half, col = l16 ----
#pragma unroll
  for (int nt = 0; nt < 4; ++nt) {
    int n = n0 + nt * 16 + l16;
    if (n >= N) continue;
    float bv = bias ? bias[n] : 0.0f;
#pragma unroll
    for (int mt = 0; mt < 2; ++mt) {
#pragma unroll
      for (int rr = 0; rr < 8; ++rr) {
        int m = m0 + wave * 32 + mt * 16 + rr + half * 8;
        if (m >= M) continue;
        float v = acc[mt][nt][rr] + bv;
        if (act == 1) v = 0.5f * v * (1.0f + erff(v * 0.70710678118654752f));
        if (C)  C[(long)m * ldc + n] = v;
        if (C2) C2[(long)m * ldc2 + n] = v + R[(long)m * ldr + n];
      }
    }
  }
}

// ---------------------------------------------------------------------------
// Flash-style WMMA attention. dk = 64 fixed. Each wave owns a 16-row q tile,
// all 4 waves of a block share (b,h) so K/V chunks are staged cooperatively.
// Online softmax; scores = (Q@K^T)*scale + bias(b,h,q,k).
// LDS rows padded (QKP=72 / VPP=40) for conflict-free b128 fragment reads.
// ---------------------------------------------------------------------------
#define QKP 72
#define VPP 40

__global__ __launch_bounds__(128) void attn_kernel(
    const float* __restrict__ Q, long q_row, long q_bat,
    const float* __restrict__ Kp, long k_row, long k_bat,
    const float* __restrict__ Vp, long v_row, long v_bat,
    float* __restrict__ O, long o_row, long o_bat,
    const float* __restrict__ biasb,
    int Sq, int Sk, float scale, int NHh)
{
  __shared__ unsigned short Qs[4][16 * QKP];  // per-wave q tile (bf16)
  __shared__ unsigned short Ks[32 * QKP];     // K chunk  [s][d]
  __shared__ unsigned short Vt[64 * VPP];     // V^T chunk [d][s]
  __shared__ unsigned short Ps[4][16 * VPP];  // per-wave probs (A layout source)

  const int t    = threadIdx.x;
  const int wave = t >> 5;
  const int lane = t & 31;
  const int half = lane >> 4;
  const int l16  = lane & 15;
  const int h = blockIdx.y;
  const int b = blockIdx.z;
  const int q0 = (blockIdx.x * 4 + wave) * 16;

  const float* Qb = Q + (long)b * q_bat + (long)h * 64;
  const float* Kb = Kp + (long)b * k_bat + (long)h * 64;
  const float* Vb = Vp + (long)b * v_bat + (long)h * 64;
  float*       Ob = O + (long)b * o_bat + (long)h * 64;

  // stage this wave's Q tile (16 x 64)
  {
    int row = lane >> 1, dseg = (lane & 1) * 32;
    int s = q0 + row;
    unsigned short* dst = &Qs[wave][row * QKP + dseg];
    if (s < Sq) {
      const float* src = Qb + (long)s * q_row + dseg;
#pragma unroll
      for (int i = 0; i < 32; ++i) dst[i] = f2bf(src[i]);
    } else {
#pragma unroll
      for (int i = 0; i < 32; ++i) dst[i] = 0;
    }
  }

  float rmax[8], rsum[8];
#pragma unroll
  for (int i = 0; i < 8; ++i) { rmax[i] = -INFINITY; rsum[i] = 0.0f; }
  v8f oacc[4] = {};

  for (int kc = 0; kc < Sk; kc += 32) {
    // cooperative stage: Ks[s][d] and Vt[d][s]
    {
      int sl = t >> 2, dseg = (t & 3) * 16;
      int s = kc + sl;
      bool val = s < Sk;
      const float* ks = Kb + (long)s * k_row + dseg;
      const float* vs = Vb + (long)s * v_row + dseg;
#pragma unroll
      for (int i = 0; i < 16; ++i) {
        Ks[sl * QKP + dseg + i]  = val ? f2bf(ks[i]) : (unsigned short)0;
        Vt[(dseg + i) * VPP + sl] = val ? f2bf(vs[i]) : (unsigned short)0;
      }
    }
    __syncthreads();

    // scores: S[16x32] = Q[16x64] @ K^T[64x32]   (2 K-dim steps, 2 n tiles)
    v8f sacc[2] = {};
#pragma unroll
    for (int kk = 0; kk < 2; ++kk) {
      Frag a;
      const unsigned int* ap =
          (const unsigned int*)(&Qs[wave][l16 * QKP + kk * 32]);
#pragma unroll
      for (int j = 0; j < 4; ++j) {
        a.u[j]     = ap[half * 4 + j];
        a.u[4 + j] = ap[8 + half * 4 + j];
      }
#pragma unroll
      for (int nt = 0; nt < 2; ++nt) {
        Frag bb;
        const unsigned int* bp =
            (const unsigned int*)(&Ks[(nt * 16 + l16) * QKP + kk * 32]);
#pragma unroll
        for (int j = 0; j < 8; ++j) bb.u[j] = bp[(j & 4) * 2 + half * 4 + (j & 3)];
        sacc[nt] = __builtin_amdgcn_wmma_f32_16x16x32_bf16(
            false, a.v, false, bb.v, (short)0, sacc[nt], false, false);
      }
    }

    // online softmax update (rows live at rr + 8*half; cols across 16 lanes)
    const int n0k = kc + l16;
    const int n1k = kc + 16 + l16;
#pragma unroll
    for (int rr = 0; rr < 8; ++rr) {
      int m = q0 + rr + half * 8;
      float b0 = 0.0f, b1 = 0.0f;
      if (biasb != nullptr && m < Sq) {
        const float* bp = biasb + ((long)(b * NHh + h) * Sq + m) * (long)Sk;
        if (n0k < Sk) b0 = bp[n0k];
        if (n1k < Sk) b1 = bp[n1k];
      }
      float s0 = (n0k < Sk) ? sacc[0][rr] * scale + b0 : -INFINITY;
      float s1 = (n1k < Sk) ? sacc[1][rr] * scale + b1 : -INFINITY;
      float mx = fmaxf(s0, s1);
      mx = fmaxf(mx, __shfl_xor(mx, 1, 32));
      mx = fmaxf(mx, __shfl_xor(mx, 2, 32));
      mx = fmaxf(mx, __shfl_xor(mx, 4, 32));
      mx = fmaxf(mx, __shfl_xor(mx, 8, 32));
      float mnew = fmaxf(rmax[rr], mx);
      float corr = __expf(rmax[rr] - mnew);
      float p0 = __expf(s0 - mnew);
      float p1 = __expf(s1 - mnew);
      float ps = p0 + p1;
      ps += __shfl_xor(ps, 1, 32);
      ps += __shfl_xor(ps, 2, 32);
      ps += __shfl_xor(ps, 4, 32);
      ps += __shfl_xor(ps, 8, 32);
      rsum[rr] = rsum[rr] * corr + ps;
      rmax[rr] = mnew;
#pragma unroll
      for (int ot = 0; ot < 4; ++ot) oacc[ot][rr] *= corr;
      Ps[wave][(rr + half * 8) * VPP + l16]      = f2bf(p0);
      Ps[wave][(rr + half * 8) * VPP + 16 + l16] = f2bf(p1);
    }

    // O += P[16x32] @ V[32x64]
    {
      Frag pa;
      const unsigned int* pp = (const unsigned int*)(&Ps[wave][l16 * VPP]);
#pragma unroll
      for (int j = 0; j < 4; ++j) {
        pa.u[j]     = pp[half * 4 + j];
        pa.u[4 + j] = pp[8 + half * 4 + j];
      }
#pragma unroll
      for (int ot = 0; ot < 4; ++ot) {
        Frag bb;
        const unsigned int* bp =
            (const unsigned int*)(&Vt[(ot * 16 + l16) * VPP]);
#pragma unroll
        for (int j = 0; j < 8; ++j) bb.u[j] = bp[(j & 4) * 2 + half * 4 + (j & 3)];
        oacc[ot] = __builtin_amdgcn_wmma_f32_16x16x32_bf16(
            false, pa.v, false, bb.v, (short)0, oacc[ot], false, false);
      }
    }
    __syncthreads();
  }

  // write O / l
#pragma unroll
  for (int rr = 0; rr < 8; ++rr) {
    int m = q0 + rr + half * 8;
    if (m >= Sq) continue;
    float inv = 1.0f / rsum[rr];
#pragma unroll
    for (int ot = 0; ot < 4; ++ot) {
      Ob[(long)m * o_row + ot * 16 + l16] = oacc[ot][rr] * inv;
    }
  }
}

// ---------------------------------------------------------------------------
// Weight transpose+convert: W_f32[K][N] -> Wt_bf16[N][K]
// ---------------------------------------------------------------------------
__global__ void tconv_kernel(const float* __restrict__ W,
                             unsigned short* __restrict__ Wt, int K, int N) {
  long idx = (long)blockIdx.x * blockDim.x + threadIdx.x;
  if (idx >= (long)K * N) return;
  int k = (int)(idx % K);
  int n = (int)(idx / K);
  Wt[idx] = f2bf(W[(long)k * N + n]);
}

// ---------------------------------------------------------------------------
// Attention-bias einsum: out[b,h,q,k] = dot(ab[b,q,k,:], Wb[:,h]) + bb[h]
// ab = attn_bias for q<N && k<N else graph_vd (virtual node edges, G=1).
// ---------------------------------------------------------------------------
__global__ void bias_kernel(const float* __restrict__ attn_bias,
                            const float* __restrict__ graph_vd,
                            const float* __restrict__ Wb,   // [8][8] (d, h)
                            const float* __restrict__ bb,   // [8]
                            float* __restrict__ out,        // [B,8,S,S]
                            int Bn, int Nn, int Sn) {
  long idx = (long)blockIdx.x * blockDim.x + threadIdx.x;
  long total = (long)Bn * Sn * Sn;
  if (idx >= total) return;
  int k = (int)(idx % Sn);
  long t2 = idx / Sn;
  int q = (int)(t2 % Sn);
  int b = (int)(t2 / Sn);
  float ab[8];
  if (q < Nn && k < Nn) {
    const float* p = attn_bias + (((long)b * Nn + q) * Nn + k) * 8;
#pragma unroll
    for (int d = 0; d < 8; ++d) ab[d] = p[d];
  } else {
#pragma unroll
    for (int d = 0; d < 8; ++d) ab[d] = graph_vd[d];
  }
#pragma unroll
  for (int hh = 0; hh < 8; ++hh) {
    float s = bb[hh];
#pragma unroll
    for (int d = 0; d < 8; ++d) s += ab[d] * Wb[d * 8 + hh];
    out[(((long)b * 8 + hh) * Sn + q) * Sn + k] = s;
  }
}

// ---------------------------------------------------------------------------
// LayerNorm over (optionally concatenated) row: wave per row.
// ---------------------------------------------------------------------------
__global__ __launch_bounds__(256) void ln_kernel(
    const float* __restrict__ in1, int D1, const float* __restrict__ in2, int D2,
    const float* __restrict__ g, const float* __restrict__ bta,
    float* __restrict__ out, int rows) {
  int wave = threadIdx.x >> 5, lane = threadIdx.x & 31;
  int row = blockIdx.x * 8 + wave;
  if (row >= rows) return;
  int D = D1 + D2;
  const float* p1 = in1 + (long)row * D1;
  const float* p2 = in2 ? in2 + (long)row * D2 : nullptr;
  float s = 0.0f, ss = 0.0f;
  for (int i = lane; i < D; i += 32) {
    float v = (i < D1) ? p1[i] : p2[i - D1];
    s += v; ss += v * v;
  }
#pragma unroll
  for (int m = 16; m >= 1; m >>= 1) {
    s  += __shfl_xor(s, m, 32);
    ss += __shfl_xor(ss, m, 32);
  }
  float mean = s / D;
  float var = ss / D - mean * mean;
  float rstd = rsqrtf(var + 1e-5f);
  float* o = out + (long)row * D;
  for (int i = lane; i < D; i += 32) {
    float v = (i < D1) ? p1[i] : p2[i - D1];
    o[i] = (v - mean) * rstd * g[i] + bta[i];
  }
}

// ---------------------------------------------------------------------------
// h = concat(x @ node_W + node_b, graph_token) along sequence dim
// ---------------------------------------------------------------------------
__global__ void build_h_kernel(const float* __restrict__ henc,
                               const float* __restrict__ gtok,
                               float* __restrict__ h,
                               int Bn, int Nn, int Sn, int Hn) {
  long idx = (long)blockIdx.x * blockDim.x + threadIdx.x;
  if (idx >= (long)Bn * Sn * Hn) return;
  int c = (int)(idx % Hn);
  long t2 = idx / Hn;
  int s = (int)(t2 % Sn);
  int b = (int)(t2 / Sn);
  h[idx] = (s < Nn) ? henc[((long)b * Nn + s) * Hn + c] : gtok[c];
}

// ---------------------------------------------------------------------------
// Final: LN of token-0 rows, 512x10 projection, log_softmax. Wave per batch.
// ---------------------------------------------------------------------------
__global__ __launch_bounds__(256) void head_kernel(
    const float* __restrict__ act, const float* __restrict__ fg,
    const float* __restrict__ fb, const float* __restrict__ Wo,
    const float* __restrict__ bo, float* __restrict__ dout,
    int Bn, int Sn, int Hn) {
  int bw = threadIdx.x >> 5, lane = threadIdx.x & 31;
  if (bw >= Bn) return;
  const float* row = act + (long)bw * Sn * Hn;
  float s = 0.0f, ss = 0.0f;
  for (int i = lane; i < Hn; i += 32) { float v = row[i]; s += v; ss += v * v; }
#pragma unroll
  for (int m = 16; m >= 1; m >>= 1) {
    s  += __shfl_xor(s, m, 32);
    ss += __shfl_xor(ss, m, 32);
  }
  float mean = s / Hn;
  float rstd = rsqrtf(ss / Hn - mean * mean + 1e-5f);
  float lg[10];
#pragma unroll
  for (int j = 0; j < 10; ++j) lg[j] = 0.0f;
  for (int i = lane; i < Hn; i += 32) {
    float nv = (row[i] - mean) * rstd * fg[i] + fb[i];
#pragma unroll
    for (int j = 0; j < 10; ++j) lg[j] += nv * Wo[i * 10 + j];
  }
#pragma unroll
  for (int j = 0; j < 10; ++j) {
#pragma unroll
    for (int m = 16; m >= 1; m >>= 1) lg[j] += __shfl_xor(lg[j], m, 32);
    lg[j] += bo[j];
  }
  float mx = lg[0];
#pragma unroll
  for (int j = 1; j < 10; ++j) mx = fmaxf(mx, lg[j]);
  float se = 0.0f;
#pragma unroll
  for (int j = 0; j < 10; ++j) se += __expf(lg[j] - mx);
  float lse = mx + __logf(se);
  if (lane < 10) dout[bw * 10 + lane] = lg[lane] - lse;
}

// ---------------------------------------------------------------------------
// Host orchestration
// ---------------------------------------------------------------------------
extern "C" void kernel_launch(void* const* d_in, const int* in_sizes, int n_in,
                              void* d_out, int out_size, void* d_ws, size_t ws_size,
                              hipStream_t stream) {
  (void)in_sizes; (void)n_in; (void)out_size; (void)ws_size;

  const int Bn = 8, Nn = 512, Sn = 513, Hn = 512, NHn = 8, Fn = 2048, Cn = 128;
  const int Ln = 6;
  const int Mr = Bn * Sn;  // 4104 activation rows
  const float scale = 0.125f;  // dk=64 -> 1/sqrt(64)

  const float* x         = (const float*)d_in[0];
  const float* attn_bias = (const float*)d_in[1];
  const float* node_W    = (const float*)d_in[2];
  const float* node_b    = (const float*)d_in[3];
  const float* graph_tok = (const float*)d_in[4];
  const float* graph_vd  = (const float*)d_in[5];
  const float* centroids = (const float*)d_in[6];
  const float* ln1_g     = (const float*)d_in[7];
  const float* ln1_b     = (const float*)d_in[8];
  const float* Wqkv_loc  = (const float*)d_in[9];
  const float* bqkv_loc  = (const float*)d_in[10];
  const float* Wbias     = (const float*)d_in[11];
  const float* bbias     = (const float*)d_in[12];
  const float* Wo_loc    = (const float*)d_in[13];
  const float* bo_loc    = (const float*)d_in[14];
  const float* Wqkv_glb  = (const float*)d_in[15];
  const float* bqkv_glb  = (const float*)d_in[16];
  const float* Wo_glb    = (const float*)d_in[17];
  const float* bo_glb    = (const float*)d_in[18];
  const float* ffn_ln_g  = (const float*)d_in[19];
  const float* ffn_ln_b  = (const float*)d_in[20];
  const float* W1        = (const float*)d_in[21];
  const float* b1        = (const float*)d_in[22];
  const float* W2        = (const float*)d_in[23];
  const float* b2        = (const float*)d_in[24];
  const float* final_g   = (const float*)d_in[25];
  const float* final_b   = (const float*)d_in[26];
  const float* out_W     = (const float*)d_in[27];
  const float* out_b     = (const float*)d_in[28];

  size_t off = 0;
  auto alloc = [&](size_t bytes) -> void* {
    void* p = (char*)d_ws + off;
    off += (bytes + 255) & ~(size_t)255;
    return p;
  };
  const size_t HH = (size_t)Hn * Hn;

  unsigned short* nodeT = (unsigned short*)alloc(2 * (size_t)Hn * 128);
  unsigned short* qkvT  = (unsigned short*)alloc(2 * (size_t)Ln * 3 * HH);
  unsigned short* oT    = (unsigned short*)alloc(2 * (size_t)Ln * HH);
  unsigned short* qTg   = (unsigned short*)alloc(2 * (size_t)Ln * HH);
  unsigned short* kvTg  = (unsigned short*)alloc(2 * (size_t)Ln * 2 * HH);
  unsigned short* oTg   = (unsigned short*)alloc(2 * (size_t)Ln * HH);
  unsigned short* w1T   = (unsigned short*)alloc(2 * (size_t)Ln * Fn * 2 * Hn);
  unsigned short* w2T   = (unsigned short*)alloc(2 * (size_t)Ln * Hn * Fn);

  float* hencb   = (float*)alloc(4 * (size_t)Bn * Nn * Hn);
  float* hbuf    = (float*)alloc(4 * (size_t)Mr * Hn);
  float* ybuf    = (float*)alloc(4 * (size_t)Mr * Hn);
  float* qkvbuf  = (float*)alloc(4 * (size_t)Mr * 3 * Hn);
  float* attnbuf = (float*)alloc(4 * (size_t)Mr * Hn);
  float* yatt    = (float*)alloc(4 * (size_t)Mr * Hn);
  float* xrbuf   = (float*)alloc(4 * (size_t)Mr * Hn);
  float* qgbuf   = (float*)alloc(4 * (size_t)Mr * Hn);
  float* kvg     = (float*)alloc(4 * (size_t)Cn * 2 * Hn);
  float* gatt    = (float*)alloc(4 * (size_t)Mr * Hn);
  float* gbuf    = (float*)alloc(4 * (size_t)Mr * Hn);
  float* y2buf   = (float*)alloc(4 * (size_t)Mr * 2 * Hn);
  float* midbuf  = (float*)alloc(4 * (size_t)Mr * Fn);
  float* biasbuf = (float*)alloc(4 * (size_t)Bn * NHn * Sn * Sn);

  auto tconv = [&](const float* W, unsigned short* Wt, int K, int N) {
    long total = (long)K * N;
    tconv_kernel<<<dim3((unsigned)((total + 255) / 256)), dim3(256), 0, stream>>>(
        W, Wt, K, N);
  };
  auto gemm = [&](const float* A, long lda, const unsigned short* Bt,
                  const float* bias, float* Cp, long ldc, float* C2, long ldc2,
                  const float* R, long ldr, int Mm, int Nm, int Km, int act) {
    dim3 grid((unsigned)((Nm + 63) / 64), (unsigned)((Mm + 127) / 128));
    gemm_bf16_kernel<<<grid, dim3(128), 0, stream>>>(
        A, lda, Bt, bias, Cp, ldc, C2, ldc2, R, ldr, Mm, Nm, Km, act);
  };

  // ---- one-time weight conversion (idempotent; safe under graph replay) ----
  tconv(node_W, nodeT, 128, Hn);
  for (int l = 0; l < Ln; ++l) {
    for (int j = 0; j < 3; ++j)
      tconv(Wqkv_loc + ((size_t)l * 3 + j) * HH, qkvT + ((size_t)l * 3 + j) * HH, Hn, Hn);
    tconv(Wo_loc + (size_t)l * HH, oT + (size_t)l * HH, Hn, Hn);
    tconv(Wqkv_glb + ((size_t)l * 3 + 0) * HH, qTg + (size_t)l * HH, Hn, Hn);
    tconv(Wqkv_glb + ((size_t)l * 3 + 1) * HH, kvTg + ((size_t)l * 2 + 0) * HH, Hn, Hn);
    tconv(Wqkv_glb + ((size_t)l * 3 + 2) * HH, kvTg + ((size_t)l * 2 + 1) * HH, Hn, Hn);
    tconv(Wo_glb + (size_t)l * HH, oTg + (size_t)l * HH, Hn, Hn);
    tconv(W1 + (size_t)l * 2 * Hn * Fn, w1T + (size_t)l * Fn * 2 * Hn, 2 * Hn, Fn);
    tconv(W2 + (size_t)l * Fn * Hn, w2T + (size_t)l * Hn * Fn, Fn, Hn);
  }

  // ---- node encoder + graph token concat ----
  gemm(x, 128, nodeT, node_b, hencb, Hn, nullptr, 0, nullptr, 0, Bn * Nn, Hn, 128, 0);
  {
    long total = (long)Bn * Sn * Hn;
    build_h_kernel<<<dim3((unsigned)((total + 255) / 256)), dim3(256), 0, stream>>>(
        hencb, graph_tok, hbuf, Bn, Nn, Sn, Hn);
  }

  const unsigned qTiles = (unsigned)(((Sn + 15) / 16 + 3) / 4);  // 9

  for (int l = 0; l < Ln; ++l) {
    // bias_l = einsum(ab, Wbias[l]) + bbias[l]
    {
      long total = (long)Bn * Sn * Sn;
      bias_kernel<<<dim3((unsigned)((total + 255) / 256)), dim3(256), 0, stream>>>(
          attn_bias, graph_vd, Wbias + (size_t)l * 64, bbias + (size_t)l * 8,
          biasbuf, Bn, Nn, Sn);
    }
    // y = LN1(out)
    ln_kernel<<<dim3((unsigned)((Mr + 7) / 8)), dim3(256), 0, stream>>>(
        hbuf, Hn, nullptr, 0, ln1_g + (size_t)l * Hn, ln1_b + (size_t)l * Hn, ybuf, Mr);
    // fused QKV projection (N = 1536)
    gemm(ybuf, Hn, qkvT + (size_t)l * 3 * HH, bqkv_loc + (size_t)l * 3 * Hn,
         qkvbuf, 3 * Hn, nullptr, 0, nullptr, 0, Mr, 3 * Hn, Hn, 0);
    // local attention with bias
    attn_kernel<<<dim3(qTiles, NHn, Bn), dim3(128), 0, stream>>>(
        qkvbuf, 3 * Hn, (long)Sn * 3 * Hn,
        qkvbuf + Hn, 3 * Hn, (long)Sn * 3 * Hn,
        qkvbuf + 2 * Hn, 3 * Hn, (long)Sn * 3 * Hn,
        attnbuf, Hn, (long)Sn * Hn, biasbuf, Sn, Sn, scale, NHn);
    // y = attn @ Wo + bo ; xr = out + y
    gemm(attnbuf, Hn, oT + (size_t)l * HH, bo_loc + (size_t)l * Hn,
         yatt, Hn, xrbuf, Hn, hbuf, Hn, Mr, Hn, Hn, 0);
    // global branch: q = y @ Wq_glb ; kv = centroids @ [Wk_glb|Wv_glb]
    gemm(yatt, Hn, qTg + (size_t)l * HH, bqkv_glb + (size_t)l * 3 * Hn,
         qgbuf, Hn, nullptr, 0, nullptr, 0, Mr, Hn, Hn, 0);
    gemm(centroids, Hn, kvTg + (size_t)l * 2 * HH,
         bqkv_glb + (size_t)l * 3 * Hn + Hn,
         kvg, 2 * Hn, nullptr, 0, nullptr, 0, Cn, 2 * Hn, Hn, 0);
    attn_kernel<<<dim3(qTiles, NHn, Bn), dim3(128), 0, stream>>>(
        qgbuf, Hn, (long)Sn * Hn,
        kvg, 2 * Hn, 0L,
        kvg + Hn, 2 * Hn, 0L,
        gatt, Hn, (long)Sn * Hn, nullptr, Sn, Cn, scale, NHn);
    // g = xr + gatt @ Wo_glb + bo_glb
    gemm(gatt, Hn, oTg + (size_t)l * HH, bo_glb + (size_t)l * Hn,
         nullptr, 0, gbuf, Hn, xrbuf, Hn, Mr, Hn, Hn, 0);
    // y2 = LN(concat(xr, g))
    ln_kernel<<<dim3((unsigned)((Mr + 7) / 8)), dim3(256), 0, stream>>>(
        xrbuf, Hn, gbuf, Hn, ffn_ln_g + (size_t)l * 2 * Hn,
        ffn_ln_b + (size_t)l * 2 * Hn, y2buf, Mr);
    // FFN: gelu(y2 @ W1 + b1) @ W2 + b2 ; out = xr + ffn
    gemm(y2buf, 2 * Hn, w1T + (size_t)l * Fn * 2 * Hn, b1 + (size_t)l * Fn,
         midbuf, Fn, nullptr, 0, nullptr, 0, Mr, Fn, 2 * Hn, 1);
    gemm(midbuf, Fn, w2T + (size_t)l * Hn * Fn, b2 + (size_t)l * Hn,
         nullptr, 0, hbuf, Hn, xrbuf, Hn, Mr, Hn, Fn, 0);
  }

  // final LN (token 0 only) + projection + log_softmax
  head_kernel<<<dim3(1), dim3(256), 0, stream>>>(
      hbuf, final_g, final_b, out_W, out_b, (float*)d_out, Bn, Sn, Hn);
}
